// Attention_64132451664201
// MI455X (gfx1250) — matrix operand order repaired and checked by reading, not compiled, not run on hardware
//
#include <hip/hip_runtime.h>
#include <hip/hip_bf16.h>
#include <stddef.h>
#include <stdint.h>

// ---------------------------------------------------------------------------
// MI455X (gfx1250) attention layer, bf16 WMMA path + async global->LDS DMA.
// Pipeline:
//   1. x (f32)            -> xb (bf16)
//   2. wq/wk/wv/wo (f32)  -> transposed bf16 (N-major, K contiguous per row)
//   3. GEMM  xb @ w{q,k,v}T -> q/k/v (f32)        [v_wmma_f32_16x16x32_bf16]
//   4. RoPE + convert      -> qb, kb (bf16)
//   5. v transpose+convert -> vTb (bf16, [kv][hd][seq])
//   6. flash attention     -> ab (bf16)           [WMMA QK^T and PV]
//   7. GEMM  ab @ woT      -> out (f32)           [WMMA]
// GEMM + attention double-buffer their LDS tiles; staging uses
// GLOBAL_LOAD_ASYNC_TO_LDS_B128 (ASYNCcnt) when the builtin is available.
// ---------------------------------------------------------------------------

#define SEQ   4096
#define DIM   2048
#define NH    16
#define NKV   4
#define HD    128

typedef __attribute__((ext_vector_type(16))) __bf16        v16bf;
typedef __attribute__((ext_vector_type(8)))  float         v8f;
typedef __attribute__((ext_vector_type(4)))  unsigned int  v4u;
typedef __attribute__((ext_vector_type(4)))  int           v4i;

#if defined(__AMDGCN__) && \
    __has_builtin(__builtin_amdgcn_global_load_async_to_lds_b128) && \
    __has_builtin(__builtin_amdgcn_s_wait_asynccnt)
#define ASYNC_LDS 1
#else
#define ASYNC_LDS 0
#endif

// D = A*B + C, bf16 inputs, f32 accumulate (16x16x32)
static __device__ __forceinline__ v8f wmma_bf16(v16bf a, v16bf b, v8f c) {
    return __builtin_amdgcn_wmma_f32_16x16x32_bf16(
        /*neg_a=*/false, a, /*neg_b=*/false, b,
        /*c_mod=*/(short)0, c, /*reuse_a=*/false, /*reuse_b=*/false);
}

// Copy NBYTES (multiple of 16) from global to LDS. Async DMA when available.
template <int NBYTES>
static __device__ __forceinline__ void stage_lds(__bf16* lds, const __bf16* g) {
#if ASYNC_LDS
#pragma unroll
    for (int i = 0; i < NBYTES / 16; ++i)
        __builtin_amdgcn_global_load_async_to_lds_b128(
            (__attribute__((address_space(1))) v4i*)(uintptr_t)((const char*)g + i * 16),
            (__attribute__((address_space(3))) v4i*)(unsigned)(uintptr_t)((char*)lds + i * 16),
            0, 0);
#else
    const v4u* s = (const v4u*)g;
    v4u*       d = (v4u*)lds;
#pragma unroll
    for (int i = 0; i < NBYTES / 16; ++i) d[i] = s[i];
#endif
}

static __device__ __forceinline__ void wait_async0() {
#if ASYNC_LDS
    __builtin_amdgcn_s_wait_asynccnt(0);
#endif
}

// Load a 16-element bf16 fragment as two 16-byte chunks (c0, c1) from a
// 64-byte (32 bf16) row segment. Works for global or LDS pointers.
static __device__ __forceinline__ v16bf frag_ld(const __bf16* base, int c0, int c1) {
    union { v16bf b; v4u u[2]; } f;
    const v4u* p = (const v4u*)base;
    f.u[0] = p[c0];
    f.u[1] = p[c1];
    return f.b;
}

// A-fragment (16x32, M x K): lanes<16 use chunks {0,2}, lanes>=16 use {1,3}.
static __device__ __forceinline__ v16bf frag_ld_A(const __bf16* row, int lh) {
    return frag_ld(row, lh ? 1 : 0, lh ? 3 : 2);
}
// B-fragment (32x16, K x N) from N-major storage: lanes<16 -> {0,2}, else {1,3}.
static __device__ __forceinline__ v16bf frag_ld_B(const __bf16* row, int lh) {
    return frag_ld(row, lh ? 1 : 0, lh ? 3 : 2);
}

// ---------------------------------------------------------------------------
// 1. flat f32 -> bf16 convert
// ---------------------------------------------------------------------------
__global__ void convert_f32_bf16(const float* __restrict__ in,
                                 __bf16* __restrict__ out, int n) {
    int i = blockIdx.x * blockDim.x + threadIdx.x;
    if (i < n) out[i] = (__bf16)in[i];
}

// ---------------------------------------------------------------------------
// 2. transpose + convert: in f32 [K][N]  ->  out bf16 [N][K]
// ---------------------------------------------------------------------------
__global__ void transpose_convert(const float* __restrict__ in,
                                  __bf16* __restrict__ out, int K, int N) {
    int i = blockIdx.x * blockDim.x + threadIdx.x;
    if (i >= K * N) return;
    int k = i / N;
    int n = i - k * N;
    out[(size_t)n * K + k] = (__bf16)in[i];
}

// ---------------------------------------------------------------------------
// 3./7. bf16 GEMM: C[M][N] (f32) = A[M][K] (bf16) * BT[N][K] (bf16)
//   128x128 block tile, 256 threads = 8 wave32 (4 row-waves x 2 col-waves),
//   each wave owns a 32x64 sub-tile = 2x4 WMMA tiles.
//   K stepped by 64 through double-buffered LDS with async DMA staging:
//   one barrier + one asynccnt wait per stage, next tile loads overlap WMMA.
// ---------------------------------------------------------------------------
__global__ void __launch_bounds__(256)
gemm_bf16(const __bf16* __restrict__ A, const __bf16* __restrict__ BT,
          float* __restrict__ C, int M, int N, int K) {
    __shared__ __bf16 sA[2][128 * 64];
    __shared__ __bf16 sB[2][128 * 64];

    const int m0  = blockIdx.y * 128;
    const int n0  = blockIdx.x * 128;
    const int tid = threadIdx.x;
    const int w   = tid >> 5;         // wave 0..7
    const int l   = tid & 31;         // lane
    const int lh  = l >> 4;           // lane half (0/1)
    const int ln  = l & 15;           // lane-in-half
    const int wr  = w >> 1;           // wave row 0..3  -> 32 rows each
    const int wc  = w & 1;            // wave col 0..1  -> 64 cols each

    v8f acc[2][4];
#pragma unroll
    for (int mi = 0; mi < 2; ++mi)
#pragma unroll
        for (int ni = 0; ni < 4; ++ni) acc[mi][ni] = (v8f)0.0f;

    // staging map: 256 threads cover 128 rows x 128 bytes; 64 B per thread
    const int srow = tid >> 1;        // 0..127
    const int sseg = tid & 1;         // 64-byte halves of a 128-byte row
    const __bf16* gA = A  + (size_t)(m0 + srow) * K + sseg * 32;
    const __bf16* gB = BT + (size_t)(n0 + srow) * K + sseg * 32;
    __bf16* lA0 = &sA[0][srow * 64 + sseg * 32];
    __bf16* lA1 = &sA[1][srow * 64 + sseg * 32];
    __bf16* lB0 = &sB[0][srow * 64 + sseg * 32];
    __bf16* lB1 = &sB[1][srow * 64 + sseg * 32];

    // prologue: stage k-block 0 into buffer 0
    stage_lds<64>(lA0, gA);
    stage_lds<64>(lB0, gB);

    const int nk = K >> 6;            // K / 64
    for (int kb = 0; kb < nk; ++kb) {
        wait_async0();
        __syncthreads();
        const int buf = kb & 1;

        // overlap: issue next tile's DMA into the other buffer
        if (kb + 1 < nk) {
            const size_t ko = (size_t)(kb + 1) * 64;
            stage_lds<64>(buf ? lA0 : lA1, gA + ko);
            stage_lds<64>(buf ? lB0 : lB1, gB + ko);
        }

#pragma unroll
        for (int ks = 0; ks < 2; ++ks) {
            v16bf af[2], bfr[4];
#pragma unroll
            for (int mi = 0; mi < 2; ++mi)
                af[mi] = frag_ld_A(&sA[buf][(wr * 32 + mi * 16 + ln) * 64 + ks * 32], lh);
#pragma unroll
            for (int ni = 0; ni < 4; ++ni)
                bfr[ni] = frag_ld_B(&sB[buf][(wc * 64 + ni * 16 + ln) * 64 + ks * 32], lh);
#pragma unroll
            for (int mi = 0; mi < 2; ++mi)
#pragma unroll
                for (int ni = 0; ni < 4; ++ni)
                    acc[mi][ni] = wmma_bf16(af[mi], bfr[ni], acc[mi][ni]);
        }
        // no trailing barrier: next iteration's top barrier protects the
        // buffer being overwritten (writes target the opposite buffer here).
    }

    // C fragment layout: VGPR g -> M = g + 8*lh ; N = ln
#pragma unroll
    for (int mi = 0; mi < 2; ++mi)
#pragma unroll
        for (int ni = 0; ni < 4; ++ni)
#pragma unroll
            for (int g = 0; g < 8; ++g) {
                int r = m0 + wr * 32 + mi * 16 + g + 8 * lh;
                int c = n0 + wc * 64 + ni * 16 + ln;
                C[(size_t)r * N + c] = acc[mi][ni][g];
            }
}

// ---------------------------------------------------------------------------
// 4. RoPE + f32->bf16 convert. in/out layout [SEQ][nheads*HD].
// ---------------------------------------------------------------------------
__global__ void rope_convert(const float* __restrict__ in,
                             const float* __restrict__ pcos,
                             const float* __restrict__ psin,
                             __bf16* __restrict__ out, int nheads) {
    int i = blockIdx.x * blockDim.x + threadIdx.x;
    int total = SEQ * nheads * (HD / 2);
    if (i >= total) return;
    int jp  = i & 63;                 // rotary pair 0..63
    int tmp = i >> 6;
    int h   = tmp % nheads;
    int t   = tmp / nheads;
    size_t base = (size_t)t * (nheads * HD) + h * HD + 2 * jp;
    float x1 = in[base];
    float x2 = in[base + 1];
    float c  = pcos[t * 64 + jp];
    float s  = psin[t * 64 + jp];
    out[base]     = (__bf16)(x1 * c - x2 * s);
    out[base + 1] = (__bf16)(x1 * s + x2 * c);
}

// ---------------------------------------------------------------------------
// 5. V transpose+convert: v f32 [SEQ][NKV*HD] -> vT bf16 [NKV][HD][SEQ]
// ---------------------------------------------------------------------------
__global__ void v_transpose(const float* __restrict__ v,
                            __bf16* __restrict__ vT) {
    int i = blockIdx.x * blockDim.x + threadIdx.x;
    if (i >= SEQ * NKV * HD) return;
    int t = i / (NKV * HD);
    int c = i - t * (NKV * HD);
    int h = c >> 7;
    int d = c & 127;
    vT[((size_t)h * HD + d) * SEQ + t] = (__bf16)v[i];
}

// ---------------------------------------------------------------------------
// 6. Flash attention (causal), bf16 WMMA, f32 online softmax.
//    grid = (SEQ/64, NH), block = 128 threads = 4 wave32.
//    Each wave owns 16 query rows; key blocks of 64; head_dim 128.
//    K/V tiles double-buffered: next key block DMAs while current computes.
// ---------------------------------------------------------------------------
__global__ void __launch_bounds__(128)
attn_kernel(const __bf16* __restrict__ Q,   // [SEQ][NH*HD]
            const __bf16* __restrict__ Kc,  // [SEQ][NKV*HD]
            const __bf16* __restrict__ VT,  // [NKV][HD][SEQ]
            __bf16* __restrict__ O) {       // [SEQ][NH*HD]
    __shared__ __bf16 sK[2][64 * 128];  // keys x head_dim
    __shared__ __bf16 sV[2][128 * 64];  // head_dim x keys (N-major for PV)
    __shared__ __bf16 sP[64 * 64];      // probs, rows x keys

    const int qb  = blockIdx.x;       // query block of 64 rows
    const int h   = blockIdx.y;       // head
    const int hk  = h >> 2;           // kv head (N_REP = 4)
    const int tid = threadIdx.x;
    const int w   = tid >> 5;
    const int l   = tid & 31;
    const int lh  = l >> 4;
    const int ln  = l & 15;

    // staging map for K tile: 128 threads cover 64 keys x 256 B
    const int kr = tid >> 1, kh = tid & 1;
    const __bf16* gK = Kc + (size_t)kr * (NKV * HD) + hk * HD + kh * 64;
    const __bf16* gV = VT + ((size_t)hk * HD + tid) * SEQ;

    // Q fragments for this wave's 16 rows (kept in registers, 4 K-steps)
    v16bf qf[4];
    {
        const __bf16* qrow = Q + (size_t)(qb * 64 + w * 16 + ln) * DIM + h * HD;
#pragma unroll
        for (int ks = 0; ks < 4; ++ks)
            qf[ks] = frag_ld_A(qrow + ks * 32, lh);
    }

    float o[8][8];                    // [hd N-tile][row g]
    float m_i[8], l_i[8];
#pragma unroll
    for (int g = 0; g < 8; ++g) {
        m_i[g] = -3.0e38f;
        l_i[g] = 0.0f;
#pragma unroll
        for (int n = 0; n < 8; ++n) o[n][g] = 0.0f;
    }

    const float scale = 0.08838834764831845f;   // 1/sqrt(128)
    const int rowg = qb * 64 + w * 16 + 8 * lh; // global row for g=0
    const int nblocks = qb + 1;                 // causal limit

    // prologue: stage key block 0 into buffer 0
    stage_lds<128>(&sK[0][kr * 128 + kh * 64], gK);
    stage_lds<128>(&sV[0][tid * 64], gV);

    for (int j = 0; j < nblocks; ++j) {
        wait_async0();
        __syncthreads();
        const int buf = j & 1;

        // overlap: DMA next key block while this one computes
        if (j + 1 < nblocks) {
            stage_lds<128>(&sK[buf ^ 1][kr * 128 + kh * 64],
                           gK + (size_t)(j + 1) * 64 * (NKV * HD));
            stage_lds<128>(&sV[buf ^ 1][tid * 64], gV + (size_t)(j + 1) * 64);
        }

        // ---- S = Q K^T : 4 key N-tiles, 4 K-steps over head_dim ----
        v8f s[4];
#pragma unroll
        for (int jn = 0; jn < 4; ++jn) {
            v8f c = (v8f)0.0f;
#pragma unroll
            for (int ks = 0; ks < 4; ++ks) {
                v16bf b = frag_ld_B(&sK[buf][(jn * 16 + ln) * 128 + ks * 32], lh);
                c = wmma_bf16(qf[ks], b, c);
            }
            s[jn] = c;
        }

        // ---- scale + causal mask ----
#pragma unroll
        for (int jn = 0; jn < 4; ++jn) {
            int col = j * 64 + jn * 16 + ln;
#pragma unroll
            for (int g = 0; g < 8; ++g) {
                float v = s[jn][g] * scale;
                if (col > rowg + g) v = -1.0e9f;
                s[jn][g] = v;
            }
        }

        // ---- online softmax: row max / exp / row sum ----
        float mnew[8], psum[8];
#pragma unroll
        for (int g = 0; g < 8; ++g) {
            float v = fmaxf(fmaxf(s[0][g], s[1][g]), fmaxf(s[2][g], s[3][g]));
#pragma unroll
            for (int off = 8; off > 0; off >>= 1)
                v = fmaxf(v, __shfl_xor(v, off, 32));
            mnew[g] = fmaxf(m_i[g], v);
            psum[g] = 0.0f;
        }
#pragma unroll
        for (int jn = 0; jn < 4; ++jn)
#pragma unroll
            for (int g = 0; g < 8; ++g) {
                float p = __expf(s[jn][g] - mnew[g]);
                psum[g] += p;
                sP[(w * 16 + g + 8 * lh) * 64 + jn * 16 + ln] = (__bf16)p;
            }
#pragma unroll
        for (int g = 0; g < 8; ++g) {
            float v = psum[g];
#pragma unroll
            for (int off = 8; off > 0; off >>= 1)
                v += __shfl_xor(v, off, 32);
            float corr = __expf(m_i[g] - mnew[g]);
            l_i[g] = l_i[g] * corr + v;
            m_i[g] = mnew[g];
#pragma unroll
            for (int n = 0; n < 8; ++n) o[n][g] *= corr;
        }
        __syncthreads();   // sP visible to the PV pass

        // ---- O += P V : 2 K-steps over keys, 8 N-tiles over head_dim ----
#pragma unroll
        for (int ks = 0; ks < 2; ++ks) {
            v16bf a = frag_ld_A(&sP[(w * 16 + ln) * 64 + ks * 32], lh);
#pragma unroll
            for (int n = 0; n < 8; ++n) {
                v16bf b = frag_ld_B(&sV[buf][(n * 16 + ln) * 64 + ks * 32], lh);
                v8f c;
#pragma unroll
                for (int g = 0; g < 8; ++g) c[g] = o[n][g];
                c = wmma_bf16(a, b, c);
#pragma unroll
                for (int g = 0; g < 8; ++g) o[n][g] = c[g];
            }
        }
        // no trailing barrier: next iteration's top barrier orders sP/sK/sV reuse
    }

    // ---- normalize and write bf16 attention output ----
#pragma unroll
    for (int g = 0; g < 8; ++g) {
        float inv = 1.0f / l_i[g];
        size_t r = (size_t)(rowg + g);
#pragma unroll
        for (int n = 0; n < 8; ++n)
            O[r * DIM + h * HD + n * 16 + ln] = (__bf16)(o[n][g] * inv);
    }
}

// ---------------------------------------------------------------------------
// launcher
// ---------------------------------------------------------------------------
extern "C" void kernel_launch(void* const* d_in, const int* in_sizes, int n_in,
                              void* d_out, int out_size, void* d_ws, size_t ws_size,
                              hipStream_t stream) {
    (void)in_sizes; (void)n_in; (void)out_size; (void)ws_size;

    const float* x    = (const float*)d_in[0];
    // d_in[1] = start_pos (0), d_in[4] = mask (causal, computed inline)
    const float* fcos = (const float*)d_in[2];
    const float* fsin = (const float*)d_in[3];
    const float* wq   = (const float*)d_in[5];
    const float* wk   = (const float*)d_in[6];
    const float* wv   = (const float*)d_in[7];
    const float* wo   = (const float*)d_in[8];
    float*       out  = (float*)d_out;

    char* ws = (char*)d_ws;
    size_t off = 0;
    __bf16* xb  = (__bf16*)(ws + off); off += (size_t)SEQ * DIM * 2;        // 16 MB
    __bf16* wqT = (__bf16*)(ws + off); off += (size_t)DIM * DIM * 2;        //  8 MB
    __bf16* wkT = (__bf16*)(ws + off); off += (size_t)DIM * (NKV*HD) * 2;   //  2 MB
    __bf16* wvT = (__bf16*)(ws + off); off += (size_t)DIM * (NKV*HD) * 2;   //  2 MB
    __bf16* woT = (__bf16*)(ws + off); off += (size_t)DIM * DIM * 2;        //  8 MB
    float*  qf  = (float*) (ws + off); off += (size_t)SEQ * DIM * 4;        // 32 MB
    float*  kf  = (float*) (ws + off); off += (size_t)SEQ * (NKV*HD) * 4;   //  8 MB
    float*  vf  = (float*) (ws + off); off += (size_t)SEQ * (NKV*HD) * 4;   //  8 MB
    __bf16* qb  = (__bf16*)(ws + off); off += (size_t)SEQ * DIM * 2;        // 16 MB
    __bf16* kb  = (__bf16*)(ws + off); off += (size_t)SEQ * (NKV*HD) * 2;   //  4 MB
    __bf16* vTb = (__bf16*)(ws + off); off += (size_t)SEQ * (NKV*HD) * 2;   //  4 MB
    __bf16* ab  = (__bf16*)(ws + off); off += (size_t)SEQ * DIM * 2;        // 16 MB

    const int T = 256;

    // 1. x -> bf16
    convert_f32_bf16<<<(SEQ * DIM + T - 1) / T, T, 0, stream>>>(x, xb, SEQ * DIM);

    // 2. weight transpose+convert ([K][N] -> [N][K] bf16)
    transpose_convert<<<(DIM * DIM + T - 1) / T, T, 0, stream>>>(wq, wqT, DIM, DIM);
    transpose_convert<<<(DIM * NKV * HD + T - 1) / T, T, 0, stream>>>(wk, wkT, DIM, NKV * HD);
    transpose_convert<<<(DIM * NKV * HD + T - 1) / T, T, 0, stream>>>(wv, wvT, DIM, NKV * HD);
    transpose_convert<<<(DIM * DIM + T - 1) / T, T, 0, stream>>>(wo, woT, DIM, DIM);

    // 3. QKV projections (WMMA bf16 GEMMs)
    gemm_bf16<<<dim3(DIM / 128, SEQ / 128), 256, 0, stream>>>(xb, wqT, qf, SEQ, DIM, DIM);
    gemm_bf16<<<dim3((NKV * HD) / 128, SEQ / 128), 256, 0, stream>>>(xb, wkT, kf, SEQ, NKV * HD, DIM);
    gemm_bf16<<<dim3((NKV * HD) / 128, SEQ / 128), 256, 0, stream>>>(xb, wvT, vf, SEQ, NKV * HD, DIM);

    // 4. RoPE + convert
    rope_convert<<<(SEQ * NH * 64 + T - 1) / T, T, 0, stream>>>(qf, fcos, fsin, qb, NH);
    rope_convert<<<(SEQ * NKV * 64 + T - 1) / T, T, 0, stream>>>(kf, fcos, fsin, kb, NKV);

    // 5. V transpose
    v_transpose<<<(SEQ * NKV * HD + T - 1) / T, T, 0, stream>>>(vf, vTb);

    // 6. flash attention (causal)
    attn_kernel<<<dim3(SEQ / 64, NH), 128, 0, stream>>>(qb, kb, vTb, ab);

    // 7. output projection
    gemm_bf16<<<dim3(DIM / 128, SEQ / 128), 256, 0, stream>>>(ab, woT, out, SEQ, DIM, DIM);
}
